// VAE_20504173871139
// MI455X (gfx1250) — compile-verified, matrix operand order checked by reading, not run on hardware
//
#include <hip/hip_runtime.h>
#include <hip/hip_bf16.h>

typedef __bf16 bf16_t;
typedef __attribute__((ext_vector_type(16))) __bf16 v16bf;
typedef __attribute__((ext_vector_type(8)))  __bf16 v8bf;
typedef __attribute__((ext_vector_type(8)))  float  v8f;

#define HDIM   1024
#define ZDIM   256
#define BATCH  64
#define TLEN   256
#define RDIM   389
#define RPAD   416
#define NPERF  300
#define NSCORE 89
#define IN0    645          // PERF + SCORE + Z
#define IN0PAD 672
#define NG     4096         // 4*H : r | z | h_n | i_n

// One 32-wide K step: shared A tile x two B (N) tiles -> two WMMAs
__device__ __forceinline__ void wmma_step(const bf16_t* __restrict__ ap,
                                          const bf16_t* __restrict__ w0p,
                                          const bf16_t* __restrict__ w1p,
                                          int koff, v8f& acc0, v8f& acc1)
{
    v8bf alo  = *(const v8bf*)(ap  + koff);
    v8bf ahi  = *(const v8bf*)(ap  + 16 + koff);
    v8bf b0lo = *(const v8bf*)(w0p + koff);
    v8bf b0hi = *(const v8bf*)(w0p + 16 + koff);
    v8bf b1lo = *(const v8bf*)(w1p + koff);
    v8bf b1hi = *(const v8bf*)(w1p + 16 + koff);
    v16bf a  = __builtin_shufflevector(alo,  ahi,  0,1,2,3,4,5,6,7,8,9,10,11,12,13,14,15);
    v16bf b0 = __builtin_shufflevector(b0lo, b0hi, 0,1,2,3,4,5,6,7,8,9,10,11,12,13,14,15);
    v16bf b1 = __builtin_shufflevector(b1lo, b1hi, 0,1,2,3,4,5,6,7,8,9,10,11,12,13,14,15);
    acc0 = __builtin_amdgcn_wmma_f32_16x16x32_bf16(false, a, false, b0,
                                                   (short)0, acc0, false, false);
    acc1 = __builtin_amdgcn_wmma_f32_16x16x32_bf16(false, a, false, b1,
                                                   (short)0, acc1, false, false);
}

// ---------------------------------------------------------------------------
// Fused GEMM:  C[m,n] = sum_k [A0|A1][m,k] * W[n,k] + bias[n]
// A0: [64 x K0] bf16 row-major, A1: [64 x K1] bf16 row-major (K1 may be 0)
// W : [N x (K0+K1)] bf16 row-major (pre-packed, zero-padded)
// C : [64 x N] f32.  Grid = N/64 blocks of 256 threads (8 wave32).
// Each wave: one 16-row M tile x two 16-col N tiles; K loop split into the
// A0 and A1 segments so the steady state is pure pointer-bump + WMMA.
// ---------------------------------------------------------------------------
__global__ __launch_bounds__(256) void gemm_bf16_wmma(
    const bf16_t* __restrict__ A0, int K0,
    const bf16_t* __restrict__ A1, int K1,
    const bf16_t* __restrict__ W,
    const float*  __restrict__ bias,
    float* __restrict__ C, int N)
{
    const int KT   = K0 + K1;
    const int lane = threadIdx.x & 31;
    const int wv   = threadIdx.x >> 5;
    const int mrow = (wv & 3) * 16;                       // M tile base (0..48)
    const int ncol = blockIdx.x * 64 + (wv >> 2) * 32;    // N tile pair base
    const int r    = lane & 15;
    const int koff = (lane < 16) ? 0 : 8;                 // 16-bit A/B lane K layout

    const bf16_t* w0 = W  + (size_t)(ncol + r)      * KT;
    const bf16_t* w1 = W  + (size_t)(ncol + 16 + r) * KT;
    const bf16_t* ap = A0 + (size_t)(mrow + r) * K0;

    v8f acc0 = {}; v8f acc1 = {};
#pragma unroll 2
    for (int kb = 0; kb < K0; kb += 32) {
        __builtin_prefetch(w0 + 256, 0, 1);               // speculative, drop-on-fault
        __builtin_prefetch(w1 + 256, 0, 1);
        wmma_step(ap, w0, w1, koff, acc0, acc1);
        ap += 32; w0 += 32; w1 += 32;
    }
    ap = A1 + (size_t)(mrow + r) * K1;
#pragma unroll 2
    for (int kb = 0; kb < K1; kb += 32) {
        __builtin_prefetch(w0 + 256, 0, 1);
        __builtin_prefetch(w1 + 256, 0, 1);
        wmma_step(ap, w0, w1, koff, acc0, acc1);
        ap += 32; w0 += 32; w1 += 32;
    }

    // C/D layout: VGPR v -> M = v + (lane>=16 ? 8 : 0), N = lane%16
    const int mb = mrow + ((lane >= 16) ? 8 : 0);
    const int n0 = ncol + r;
    const int n1 = n0 + 16;
    const float bb0 = bias[n0], bb1 = bias[n1];
#pragma unroll
    for (int v = 0; v < 8; ++v) {
        C[(size_t)(mb + v) * N + n0] = acc0[v] + bb0;
        C[(size_t)(mb + v) * N + n1] = acc1[v] + bb1;
    }
}

// ---------------------------------------------------------------------------
// f32 -> bf16 convert with zero-padding into a packed region
// ---------------------------------------------------------------------------
__global__ void cvt_pad(const float* __restrict__ src, int srcStride,
                        bf16_t* __restrict__ dst, int dstStride,
                        int rows, int kcopy, int kregion)
{
    int idx = blockIdx.x * blockDim.x + threadIdx.x;
    if (idx >= rows * kregion) return;
    int rr = idx / kregion, k = idx % kregion;
    dst[(size_t)rr * dstStride + k] =
        (k < kcopy) ? (bf16_t)src[(size_t)rr * srcStride + k] : (bf16_t)0.0f;
}

// bias for the 4H-extended GRU GEMM: [b_ih+b_hh (2H) | b_hh_n (H) | b_ih_n (H)]
__global__ void pack_gru_bias(const float* __restrict__ bih,
                              const float* __restrict__ bhh,
                              float* __restrict__ dst)
{
    int i = blockIdx.x * blockDim.x + threadIdx.x;
    if (i >= 4 * HDIM) return;
    if (i < 2 * HDIM)      dst[i] = bih[i] + bhh[i];
    else if (i < 3 * HDIM) dst[i] = bhh[i];           // h_n bias
    else                   dst[i] = bih[i - HDIM];    // i_n bias
}

__global__ void bias_pad(const float* __restrict__ s, float* __restrict__ d,
                         int nsrc, int ndst)
{
    int i = blockIdx.x * blockDim.x + threadIdx.x;
    if (i < ndst) d[i] = (i < nsrc) ? s[i] : 0.0f;
}

// GRU gate epilogue over the 64 x 4H gate buffer
__global__ void gru_gate(const float* __restrict__ G,
                         const float* __restrict__ hprev,
                         float* __restrict__ hout,
                         bf16_t* __restrict__ hbf)
{
    int i = blockIdx.x * blockDim.x + threadIdx.x;
    if (i >= BATCH * HDIM) return;
    int m = i >> 10, u = i & (HDIM - 1);
    const float* g = G + (size_t)m * NG;
    float rg = 1.0f / (1.0f + __expf(-g[u]));
    float zg = 1.0f / (1.0f + __expf(-g[HDIM + u]));
    float nn = tanhf(g[3 * HDIM + u] + rg * g[2 * HDIM + u]);
    float h  = (1.0f - zg) * nn + zg * hprev[i];
    hout[i] = h;
    hbf[i]  = (bf16_t)h;
}

// hidden = concat(h_f, h_b) / sqrt(1+eps), as bf16 GEMM operand
__global__ void build_hidden(const float* __restrict__ hf,
                             const float* __restrict__ hb,
                             bf16_t* __restrict__ dst)
{
    int i = blockIdx.x * blockDim.x + threadIdx.x;
    if (i >= BATCH * 2 * HDIM) return;
    int m = i >> 11, c = i & (2 * HDIM - 1);
    float v = (c < HDIM) ? hf[m * HDIM + c] : hb[m * HDIM + (c - HDIM)];
    dst[i] = (bf16_t)(v * 0.9999950000374997f);   // 1/sqrt(1+1e-5)
}

__global__ void tanh_init3(const float* __restrict__ g,
                           float* h0, float* h1, float* h2,
                           bf16_t* b0, bf16_t* b1, bf16_t* b2)
{
    int i = blockIdx.x * blockDim.x + threadIdx.x;
    if (i >= BATCH * HDIM) return;
    float v = tanhf(g[i]);
    h0[i] = v; h1[i] = v; h2[i] = v;
    bf16_t t = (bf16_t)v;
    b0[i] = t; b1[i] = t; b2[i] = t;
}

// decoder step input: [one_hot(argmax, 300) | score_{j-1} (89) | z (256) | pad]
__global__ void build_inp(const int* __restrict__ amax,
                          const float* __restrict__ score,  // x + (j-1)*B*R
                          const float* __restrict__ z,      // 64 x 256 f32
                          bf16_t* __restrict__ inp, int first)
{
    int idx = blockIdx.x * blockDim.x + threadIdx.x;
    if (idx >= BATCH * IN0PAD) return;
    int m = idx / IN0PAD, c = idx % IN0PAD;
    float v;
    if (c < NPERF) {
        v = (amax[m] == c) ? 1.0f : 0.0f;
    } else if (c < NPERF + NSCORE) {
        int s = c - NPERF;
        v = first ? ((s == NSCORE - 1) ? 1.0f : 0.0f)
                  : score[(size_t)m * RDIM + s];
    } else if (c < IN0) {
        v = z[m * ZDIM + (c - NPERF - NSCORE)];
    } else {
        v = 0.0f;
    }
    inp[idx] = (bf16_t)v;
}

// 3-segment log_softmax over logits[64 x 320], write outs[B,T,PERF], argmax
__global__ __launch_bounds__(128) void softmax_argmax(
    const float* __restrict__ logits, float* __restrict__ out,
    int j, int* __restrict__ amax)
{
    __shared__ float red[128];
    __shared__ float seg_ls[3];
    __shared__ float bv[128];
    __shared__ int   bi[128];
    const int m = blockIdx.x, t = threadIdx.x;
    const float* L = logits + (size_t)m * 320;
    for (int s = 0; s < 3; ++s) {
        int base = s * 100;
        float mx = -1e30f;
        for (int c = t; c < 100; c += 128) mx = fmaxf(mx, L[base + c]);
        red[t] = mx; __syncthreads();
        for (int o = 64; o > 0; o >>= 1) {
            if (t < o) red[t] = fmaxf(red[t], red[t + o]);
            __syncthreads();
        }
        mx = red[0]; __syncthreads();
        float sm = 0.0f;
        for (int c = t; c < 100; c += 128) sm += __expf(L[base + c] - mx);
        red[t] = sm; __syncthreads();
        for (int o = 64; o > 0; o >>= 1) {
            if (t < o) red[t] += red[t + o];
            __syncthreads();
        }
        if (t == 0) seg_ls[s] = mx + __logf(red[0]);
        __syncthreads();
    }
    float bestv = -1e30f; int besti = 0;
    float* orow = out + (size_t)m * TLEN * NPERF + (size_t)j * NPERF;
    for (int c = t; c < NPERF; c += 128) {
        float o = L[c] - seg_ls[c / 100];
        orow[c] = o;
        if (o > bestv) { bestv = o; besti = c; }
    }
    bv[t] = bestv; bi[t] = besti; __syncthreads();
    for (int o = 64; o > 0; o >>= 1) {
        if (t < o) {
            if (bv[t + o] > bv[t] || (bv[t + o] == bv[t] && bi[t + o] < bi[t])) {
                bv[t] = bv[t + o]; bi[t] = bi[t + o];
            }
        }
        __syncthreads();
    }
    if (t == 0) amax[m] = bi[0];
}

__global__ void fill_f32(float* p, float v, int n)
{ int i = blockIdx.x * blockDim.x + threadIdx.x; if (i < n) p[i] = v; }
__global__ void fill_bf16(bf16_t* p, float v, int n)
{ int i = blockIdx.x * blockDim.x + threadIdx.x; if (i < n) p[i] = (bf16_t)v; }
__global__ void fill_i32(int* p, int v, int n)
{ int i = blockIdx.x * blockDim.x + threadIdx.x; if (i < n) p[i] = v; }
__global__ void f2bf(const float* __restrict__ s, bf16_t* __restrict__ d, int n)
{ int i = blockIdx.x * blockDim.x + threadIdx.x; if (i < n) d[i] = (bf16_t)s[i]; }

// ---------------------------------------------------------------------------
static inline void cvt_launch(hipStream_t st, const float* src, int ss,
                              bf16_t* dst, int ds, long rows, int kcopy, int kregion)
{
    long total = rows * (long)kregion;
    int blocks = (int)((total + 255) / 256);
    cvt_pad<<<blocks, 256, 0, st>>>(src, ss, dst, ds, (int)rows, kcopy, kregion);
}

// Pack GRU weights into [4H x (1024 + kih_pad)] with N-extension for the n-gate
static void pack_gru_w(hipStream_t st, const float* w_ih, int kih, int kih_pad,
                       const float* w_hh, bf16_t* dst)
{
    const int ld = HDIM + kih_pad;
    // rows 0..2H-1 : r,z combined  ->  [w_hh | w_ih]
    cvt_launch(st, w_hh, HDIM, dst, ld, 2 * HDIM, HDIM, HDIM);
    cvt_launch(st, w_ih, kih, dst + HDIM, ld, 2 * HDIM, kih, kih_pad);
    // rows 2H..3H-1 : h_n only   ->  [w_hh_n | 0]
    cvt_launch(st, w_hh + (size_t)2 * HDIM * HDIM, HDIM,
               dst + (size_t)2 * HDIM * ld, ld, HDIM, HDIM, HDIM);
    cvt_launch(st, w_hh, HDIM, dst + (size_t)2 * HDIM * ld + HDIM, ld, HDIM, 0, kih_pad);
    // rows 3H..4H-1 : i_n only   ->  [0 | w_ih_n]
    cvt_launch(st, w_hh, HDIM, dst + (size_t)3 * HDIM * ld, ld, HDIM, 0, HDIM);
    cvt_launch(st, w_ih + (size_t)2 * HDIM * kih, kih,
               dst + (size_t)3 * HDIM * ld + HDIM, ld, HDIM, kih, kih_pad);
}

extern "C" void kernel_launch(void* const* d_in, const int* in_sizes, int n_in,
                              void* d_out_v, int out_size, void* d_ws, size_t ws_size,
                              hipStream_t stream)
{
    (void)in_sizes; (void)n_in; (void)out_size; (void)ws_size;
    const float* x      = (const float*)d_in[0];
    const float* w_ih_f = (const float*)d_in[2];
    const float* w_hh_f = (const float*)d_in[3];
    const float* b_ih_f = (const float*)d_in[4];
    const float* b_hh_f = (const float*)d_in[5];
    const float* w_ih_b = (const float*)d_in[6];
    const float* w_hh_b = (const float*)d_in[7];
    const float* b_ih_b = (const float*)d_in[8];
    const float* b_hh_b = (const float*)d_in[9];
    const float* w_mu   = (const float*)d_in[10];
    const float* b_mu   = (const float*)d_in[11];
    const float* w_init = (const float*)d_in[14];
    const float* b_init = (const float*)d_in[15];
    const float* w_ih0  = (const float*)d_in[16];
    const float* w_hh0  = (const float*)d_in[17];
    const float* b_ih0  = (const float*)d_in[18];
    const float* b_hh0  = (const float*)d_in[19];
    const float* w_ih1  = (const float*)d_in[20];
    const float* w_hh1  = (const float*)d_in[21];
    const float* b_ih1  = (const float*)d_in[22];
    const float* b_hh1  = (const float*)d_in[23];
    const float* w_ih2  = (const float*)d_in[24];
    const float* w_hh2  = (const float*)d_in[25];
    const float* b_ih2  = (const float*)d_in[26];
    const float* b_hh2  = (const float*)d_in[27];
    const float* w_out  = (const float*)d_in[28];
    const float* b_out  = (const float*)d_in[29];
    float* out = (float*)d_out_v;

    char* base = (char*)d_ws;
    size_t off = 0;
    auto alloc = [&](size_t bytes) -> void* {
        void* p = base + off;
        off = (off + bytes + 255) & ~(size_t)255;
        return p;
    };
    // packed bf16 weights
    bf16_t* Wef  = (bf16_t*)alloc((size_t)NG * 1440 * 2);
    bf16_t* Web  = (bf16_t*)alloc((size_t)NG * 1440 * 2);
    bf16_t* Wd0  = (bf16_t*)alloc((size_t)NG * 1696 * 2);
    bf16_t* Wd1  = (bf16_t*)alloc((size_t)NG * 2048 * 2);
    bf16_t* Wd2  = (bf16_t*)alloc((size_t)NG * 2048 * 2);
    bf16_t* Wmu  = (bf16_t*)alloc((size_t)ZDIM * 2048 * 2);
    bf16_t* Wini = (bf16_t*)alloc((size_t)HDIM * ZDIM * 2);
    bf16_t* Wou  = (bf16_t*)alloc((size_t)320 * HDIM * 2);
    bf16_t* Xbf  = (bf16_t*)alloc((size_t)TLEN * BATCH * RPAD * 2);
    // biases
    float* Bef = (float*)alloc(NG * 4);
    float* Beb = (float*)alloc(NG * 4);
    float* Bd0 = (float*)alloc(NG * 4);
    float* Bd1 = (float*)alloc(NG * 4);
    float* Bd2 = (float*)alloc(NG * 4);
    float* Bou = (float*)alloc(320 * 4);
    // activations
    float*  h_f    = (float*)alloc(BATCH * HDIM * 4);
    float*  h_b    = (float*)alloc(BATCH * HDIM * 4);
    float*  hx0    = (float*)alloc(BATCH * HDIM * 4);
    float*  hx1    = (float*)alloc(BATCH * HDIM * 4);
    float*  hx2    = (float*)alloc(BATCH * HDIM * 4);
    bf16_t* h_f_bf = (bf16_t*)alloc(BATCH * HDIM * 2);
    bf16_t* h_b_bf = (bf16_t*)alloc(BATCH * HDIM * 2);
    bf16_t* hx0_bf = (bf16_t*)alloc(BATCH * HDIM * 2);
    bf16_t* hx1_bf = (bf16_t*)alloc(BATCH * HDIM * 2);
    bf16_t* hx2_bf = (bf16_t*)alloc(BATCH * HDIM * 2);
    bf16_t* hidden = (bf16_t*)alloc(BATCH * 2 * HDIM * 2);
    float*  zbuf   = (float*)alloc(BATCH * ZDIM * 4);
    bf16_t* z_bf   = (bf16_t*)alloc(BATCH * ZDIM * 2);
    float*  gh     = (float*)alloc((size_t)BATCH * NG * 4);
    bf16_t* inp_bf = (bf16_t*)alloc(BATCH * IN0PAD * 2);
    float*  logits = (float*)alloc(BATCH * 320 * 4);
    int*    amax   = (int*)alloc(BATCH * 4);

    // ---- weight/bias packing (fp32 -> bf16, zero pad) ----
    pack_gru_w(stream, w_ih_f, RDIM, RPAD, w_hh_f, Wef);
    pack_gru_w(stream, w_ih_b, RDIM, RPAD, w_hh_b, Web);
    pack_gru_w(stream, w_ih0, IN0, IN0PAD, w_hh0, Wd0);
    pack_gru_w(stream, w_ih1, HDIM, HDIM, w_hh1, Wd1);
    pack_gru_w(stream, w_ih2, HDIM, HDIM, w_hh2, Wd2);
    cvt_launch(stream, w_mu, 2 * HDIM, Wmu, 2 * HDIM, ZDIM, 2 * HDIM, 2 * HDIM);
    cvt_launch(stream, w_init, ZDIM, Wini, ZDIM, HDIM, ZDIM, ZDIM);
    cvt_launch(stream, w_out, HDIM, Wou, HDIM, NPERF, HDIM, HDIM);
    cvt_launch(stream, w_out, HDIM, Wou + (size_t)NPERF * HDIM, HDIM, 20, 0, HDIM);
    cvt_launch(stream, x, RDIM, Xbf, RPAD, (long)TLEN * BATCH, RDIM, RPAD);
    pack_gru_bias<<<16, 256, 0, stream>>>(b_ih_f, b_hh_f, Bef);
    pack_gru_bias<<<16, 256, 0, stream>>>(b_ih_b, b_hh_b, Beb);
    pack_gru_bias<<<16, 256, 0, stream>>>(b_ih0, b_hh0, Bd0);
    pack_gru_bias<<<16, 256, 0, stream>>>(b_ih1, b_hh1, Bd1);
    pack_gru_bias<<<16, 256, 0, stream>>>(b_ih2, b_hh2, Bd2);
    bias_pad<<<2, 256, 0, stream>>>(b_out, Bou, NPERF, 320);

    // ---- init state ----
    fill_f32 <<<256, 256, 0, stream>>>(h_f, 0.0f, BATCH * HDIM);
    fill_f32 <<<256, 256, 0, stream>>>(h_b, 0.0f, BATCH * HDIM);
    fill_bf16<<<256, 256, 0, stream>>>(h_f_bf, 0.0f, BATCH * HDIM);
    fill_bf16<<<256, 256, 0, stream>>>(h_b_bf, 0.0f, BATCH * HDIM);
    fill_i32 <<<1, 64, 0, stream>>>(amax, NPERF - 1, BATCH);
    // outputs 2 & 3 of the reference tuple are zeros
    fill_f32 <<<128, 256, 0, stream>>>(out + (size_t)BATCH * TLEN * NPERF,
                                       0.0f, 2 * BATCH * ZDIM);

    // ---- bidirectional GRU encoder (fused gi+gh GEMM per step) ----
    for (int t = 0; t < TLEN; ++t) {
        gemm_bf16_wmma<<<NG / 64, 256, 0, stream>>>(
            h_f_bf, HDIM, Xbf + (size_t)t * BATCH * RPAD, RPAD, Wef, Bef, gh, NG);
        gru_gate<<<256, 256, 0, stream>>>(gh, h_f, h_f, h_f_bf);
    }
    for (int t = 0; t < TLEN; ++t) {
        gemm_bf16_wmma<<<NG / 64, 256, 0, stream>>>(
            h_b_bf, HDIM, Xbf + (size_t)(TLEN - 1 - t) * BATCH * RPAD, RPAD,
            Web, Beb, gh, NG);
        gru_gate<<<256, 256, 0, stream>>>(gh, h_b, h_b, h_b_bf);
    }

    // ---- latent: z = mean(hidden)   (w_var / stddev are dead code) ----
    build_hidden<<<512, 256, 0, stream>>>(h_f, h_b, hidden);
    gemm_bf16_wmma<<<ZDIM / 64, 256, 0, stream>>>(
        hidden, 2 * HDIM, hidden, 0, Wmu, b_mu, zbuf, ZDIM);
    f2bf<<<64, 256, 0, stream>>>(zbuf, z_bf, BATCH * ZDIM);

    // ---- hinit = tanh(z @ w_init.T + b_init) -> hx0=hx1=hx2 ----
    gemm_bf16_wmma<<<HDIM / 64, 256, 0, stream>>>(
        z_bf, ZDIM, z_bf, 0, Wini, b_init, gh, HDIM);
    tanh_init3<<<256, 256, 0, stream>>>(gh, hx0, hx1, hx2, hx0_bf, hx1_bf, hx2_bf);

    // ---- autoregressive 3-layer GRU decoder ----
    for (int j = 0; j < TLEN; ++j) {
        build_inp<<<(BATCH * IN0PAD + 255) / 256, 256, 0, stream>>>(
            amax, (j == 0) ? x : x + (size_t)(j - 1) * BATCH * RDIM,
            zbuf, inp_bf, (j == 0) ? 1 : 0);
        // layer 0 : A = [hx0 | inp]
        gemm_bf16_wmma<<<NG / 64, 256, 0, stream>>>(
            hx0_bf, HDIM, inp_bf, IN0PAD, Wd0, Bd0, gh, NG);
        gru_gate<<<256, 256, 0, stream>>>(gh, hx0, hx0, hx0_bf);
        // layer 1 : h_prev = (j==0 ? hx0 : hx1), x = hx0
        const float*  hp1  = (j == 0) ? hx0 : hx1;
        const bf16_t* hp1b = (j == 0) ? hx0_bf : hx1_bf;
        gemm_bf16_wmma<<<NG / 64, 256, 0, stream>>>(
            hp1b, HDIM, hx0_bf, HDIM, Wd1, Bd1, gh, NG);
        gru_gate<<<256, 256, 0, stream>>>(gh, hp1, hx1, hx1_bf);
        // layer 2 : h_prev = (j==0 ? hx1 : hx2), x = hx1
        const float*  hp2  = (j == 0) ? hx1 : hx2;
        const bf16_t* hp2b = (j == 0) ? hx1_bf : hx2_bf;
        gemm_bf16_wmma<<<NG / 64, 256, 0, stream>>>(
            hp2b, HDIM, hx1_bf, HDIM, Wd2, Bd2, gh, NG);
        gru_gate<<<256, 256, 0, stream>>>(gh, hp2, hx2, hx2_bf);
        // logits + segmented log-softmax + argmax feedback
        gemm_bf16_wmma<<<320 / 64, 256, 0, stream>>>(
            hx2_bf, HDIM, hx2_bf, 0, Wou, Bou, logits, 320);
        softmax_argmax<<<BATCH, 128, 0, stream>>>(logits, out, j, amax);
    }
}
